// SimpleExpertChoiceMoE_64536178590119
// MI455X (gfx1250) — matrix operand order repaired and checked by reading, not compiled, not run on hardware
//
#include <hip/hip_runtime.h>

// ---------------------------------------------------------------------------
// Expert-choice MoE forward for MI455X (gfx1250), wave32 + WMMA bf16.
// B=4,S=2048 -> T=8192 tokens, D=1024, H=4096, E=8, top_k=1024.
//
// GEMM tiling: block 128(M) x 64(N), 8 waves (4x2), wave tile 32x32
// (4x v_wmma_f32_16x16x32_bf16 per K-step), K-step 32, double-buffered LDS,
// v_perm_b32 bf16 packing, async global->LDS staging for GEMM2's A tiles.
// ---------------------------------------------------------------------------

typedef __attribute__((ext_vector_type(16))) __bf16 v16bf;
typedef __attribute__((ext_vector_type(8)))  float  v8f;

#define T_TOK 8192
#define DIM   1024
#define HID   4096
#define NE    8
#define TOPK  1024

#define LDSTRIDE 40   // halves per LDS tile row (80 B, 16B-aligned, bank-skewed)

#define USE_ASYNC_LDS 1

static __device__ __forceinline__ unsigned short f2bf(float f) {
    union { float f; unsigned u; } c; c.f = f;
    unsigned r = c.u + 0x7FFFu + ((c.u >> 16) & 1u);   // RNE (epilogue only)
    return (unsigned short)(r >> 16);
}
// pack bf16(lo), bf16(hi) into one dword with a single v_perm_b32 (truncation)
static __device__ __forceinline__ unsigned pkbf(float lo, float hi) {
    return __builtin_amdgcn_perm(__float_as_uint(hi), __float_as_uint(lo),
                                 0x07060302u);
}

union AFrag { uint4 q[2]; v16bf v; };

static __device__ __forceinline__ v8f wmma_bf16(const v16bf& a, const v16bf& b, v8f c) {
    return __builtin_amdgcn_wmma_f32_16x16x32_bf16(
        false, a, false, b, (short)0, c, false, false);
}

// ---------------------------------------------------------------------------
// 0) zero the scatter-add destination (results region of d_out)
// ---------------------------------------------------------------------------
__global__ void moe_zero(float* __restrict__ p) {
    size_t i = ((size_t)blockIdx.x * blockDim.x + threadIdx.x) * 4;
    *reinterpret_cast<float4*>(p + i) = make_float4(0.f, 0.f, 0.f, 0.f);
}

// ---------------------------------------------------------------------------
// 1) router: logits[T,E] = x @ rw^T ; probsT[E,T] = softmax over E
// ---------------------------------------------------------------------------
__global__ void __launch_bounds__(256) moe_router(
        const float* __restrict__ x, const float* __restrict__ rw,
        float* __restrict__ logits, float* __restrict__ probsT) {
    __shared__ float rws[NE * DIM];
    int tid = threadIdx.x;
    for (int i = tid; i < NE * DIM; i += 256) rws[i] = rw[i];
    __syncthreads();

    int wave = tid >> 5, lane = tid & 31;
    int tok  = blockIdx.x * 8 + wave;
    const float* xp = x + (size_t)tok * DIM;

    float acc[NE];
#pragma unroll
    for (int e = 0; e < NE; ++e) acc[e] = 0.f;
    for (int d = lane; d < DIM; d += 32) {
        float xv = xp[d];
#pragma unroll
        for (int e = 0; e < NE; ++e) acc[e] = fmaf(xv, rws[e * DIM + d], acc[e]);
    }
#pragma unroll
    for (int e = 0; e < NE; ++e) {
#pragma unroll
        for (int off = 16; off > 0; off >>= 1)
            acc[e] += __shfl_xor(acc[e], off, 32);
    }
    float mx = acc[0];
#pragma unroll
    for (int e = 1; e < NE; ++e) mx = fmaxf(mx, acc[e]);
    float den = 0.f;
#pragma unroll
    for (int e = 0; e < NE; ++e) den += expf(acc[e] - mx);

    if (lane < NE) {
        logits[(size_t)tok * NE + lane]    = acc[lane];
        probsT[(size_t)lane * T_TOK + tok] = expf(acc[lane] - mx) / den;
    }
}

// ---------------------------------------------------------------------------
// 2) top-k per expert: bitonic sort (descending) of 8192 (prob, idx) in LDS
// ---------------------------------------------------------------------------
__global__ void __launch_bounds__(1024) moe_topk(
        const float* __restrict__ probsT,
        float* __restrict__ gateW, int* __restrict__ selOut) {
    __shared__ float key[T_TOK];
    __shared__ int   idx[T_TOK];
    int e = blockIdx.x, tid = threadIdx.x;

    for (int i = tid; i < T_TOK; i += 1024) {
        key[i] = probsT[(size_t)e * T_TOK + i];
        idx[i] = i;
    }
    __syncthreads();

    for (unsigned k = 2; k <= T_TOK; k <<= 1) {
        for (unsigned j = k >> 1; j > 0; j >>= 1) {
            for (unsigned t = tid; t < T_TOK / 2; t += 1024) {
                unsigned i = ((t & ~(j - 1)) << 1) | (t & (j - 1));
                unsigned p = i | j;
                bool desc = ((i & k) == 0);
                float a = key[i], b = key[p];
                bool sw = desc ? (a < b) : (a > b);
                if (sw) {
                    key[i] = b; key[p] = a;
                    int ti = idx[i]; idx[i] = idx[p]; idx[p] = ti;
                }
            }
            __syncthreads();
        }
    }
    if (tid < TOPK) {
        gateW[e * TOPK + tid]  = key[tid];
        selOut[e * TOPK + tid] = idx[tid];
    }
}

// ---------------------------------------------------------------------------
// fragment assembly + 4 WMMAs from double-buffered LDS tiles
//   As: [m][k] row-major bf16, 128 rows; Bs: [n][k] transposed bf16, 64 rows
// ---------------------------------------------------------------------------
static __device__ __forceinline__ void frag_compute(
        const unsigned short* __restrict__ As,
        const unsigned short* __restrict__ Bs,
        int wm, int wn, int g, int mi, v8f acc[2][2]) {
    AFrag af[2], bf[2];
#pragma unroll
    for (int sm = 0; sm < 2; ++sm) {
        int row = wm * 32 + sm * 16 + mi;
        const uint4* ap = reinterpret_cast<const uint4*>(&As[row * LDSTRIDE + g * 8]);
        af[sm].q[0] = ap[0];
        af[sm].q[1] = ap[2];            // +32 B = k+16 run
    }
#pragma unroll
    for (int sn = 0; sn < 2; ++sn) {
        int n = wn * 32 + sn * 16 + mi;
        const uint4* bp = reinterpret_cast<const uint4*>(&Bs[n * LDSTRIDE + g * 8]);
        bf[sn].q[0] = bp[0];
        bf[sn].q[1] = bp[2];
    }
#pragma unroll
    for (int sm = 0; sm < 2; ++sm)
#pragma unroll
        for (int sn = 0; sn < 2; ++sn)
            acc[sm][sn] = wmma_bf16(af[sm].v, bf[sn].v, acc[sm][sn]);
}

// ---------------------------------------------------------------------------
// 3) GEMM1: h[e,k,:] = gelu( gather(x)[e,k,:] @ w1[e] + b1[e] ), h in bf16
// ---------------------------------------------------------------------------
__global__ void __launch_bounds__(256) moe_gemm1(
        const float* __restrict__ x, const float* __restrict__ w1,
        const float* __restrict__ b1, const int* __restrict__ sel,
        unsigned short* __restrict__ hbuf) {
    __shared__ __align__(16) unsigned short As[2][128 * LDSTRIDE];
    __shared__ __align__(16) unsigned short Bs[2][64 * LDSTRIDE];

    int e  = blockIdx.z;
    int m0 = blockIdx.x * 128;
    int n0 = blockIdx.y * 64;
    int tid = threadIdx.x;
    int wave = tid >> 5, lane = tid & 31;
    int wm = wave >> 1, wn = wave & 1;       // 4 x 2 wave grid, 32x32 per wave
    int g  = lane >> 4, mi = lane & 15;

    int lm = tid >> 1, lq = tid & 1;         // A loader: 128 rows x (16 halves)
    int n4 = tid & 15, kp = tid >> 4;        // B loader: 2 k-rows x 4 n per thread
    int tokA = sel[e * TOPK + m0 + lm];
    const float* aRow = x + (size_t)tokA * DIM;
    const float* w1e  = w1 + (size_t)e * DIM * HID;

    v8f acc[2][2] = {};
    unsigned aPk[8];
    unsigned bPk[4];

    // ---- stage K-step 0 ----
    {
        const float4* sA = reinterpret_cast<const float4*>(aRow + lq * 16);
        float4 a0 = sA[0], a1 = sA[1], a2 = sA[2], a3 = sA[3];
        aPk[0] = pkbf(a0.x, a0.y); aPk[1] = pkbf(a0.z, a0.w);
        aPk[2] = pkbf(a1.x, a1.y); aPk[3] = pkbf(a1.z, a1.w);
        aPk[4] = pkbf(a2.x, a2.y); aPk[5] = pkbf(a2.z, a2.w);
        aPk[6] = pkbf(a3.x, a3.y); aPk[7] = pkbf(a3.z, a3.w);
        const float* r0 = w1e + (size_t)(2 * kp) * HID + n0 + n4 * 4;
        float4 f0 = *reinterpret_cast<const float4*>(r0);
        float4 f1 = *reinterpret_cast<const float4*>(r0 + HID);
        bPk[0] = pkbf(f0.x, f1.x); bPk[1] = pkbf(f0.y, f1.y);
        bPk[2] = pkbf(f0.z, f1.z); bPk[3] = pkbf(f0.w, f1.w);
    }
    {
        uint4* dstA = reinterpret_cast<uint4*>(&As[0][lm * LDSTRIDE + lq * 16]);
        dstA[0] = make_uint4(aPk[0], aPk[1], aPk[2], aPk[3]);
        dstA[1] = make_uint4(aPk[4], aPk[5], aPk[6], aPk[7]);
#pragma unroll
        for (int i = 0; i < 4; ++i)
            *reinterpret_cast<unsigned*>(&Bs[0][(n4 * 4 + i) * LDSTRIDE + kp * 2]) = bPk[i];
    }
    __syncthreads();

    const int NK = DIM / 32;
    for (int kt = 0; kt < NK; ++kt) {
        int cur = kt & 1, nxt = cur ^ 1;
        bool more = (kt + 1) < NK;
        if (more) {
            int d0 = (kt + 1) * 32;
            const float4* sA = reinterpret_cast<const float4*>(aRow + d0 + lq * 16);
            float4 a0 = sA[0], a1 = sA[1], a2 = sA[2], a3 = sA[3];
            aPk[0] = pkbf(a0.x, a0.y); aPk[1] = pkbf(a0.z, a0.w);
            aPk[2] = pkbf(a1.x, a1.y); aPk[3] = pkbf(a1.z, a1.w);
            aPk[4] = pkbf(a2.x, a2.y); aPk[5] = pkbf(a2.z, a2.w);
            aPk[6] = pkbf(a3.x, a3.y); aPk[7] = pkbf(a3.z, a3.w);
            const float* r0 = w1e + (size_t)(d0 + 2 * kp) * HID + n0 + n4 * 4;
            float4 f0 = *reinterpret_cast<const float4*>(r0);
            float4 f1 = *reinterpret_cast<const float4*>(r0 + HID);
            bPk[0] = pkbf(f0.x, f1.x); bPk[1] = pkbf(f0.y, f1.y);
            bPk[2] = pkbf(f0.z, f1.z); bPk[3] = pkbf(f0.w, f1.w);
            if ((kt + 2) < NK)   // weight-stream prefetch one step further ahead
                __builtin_prefetch(w1e + (size_t)((kt + 2) * 32 + 2 * kp) * HID + n0 + n4 * 4, 0, 1);
        }
        frag_compute(As[cur], Bs[cur], wm, wn, g, mi, acc);
        if (more) {
            uint4* dstA = reinterpret_cast<uint4*>(&As[nxt][lm * LDSTRIDE + lq * 16]);
            dstA[0] = make_uint4(aPk[0], aPk[1], aPk[2], aPk[3]);
            dstA[1] = make_uint4(aPk[4], aPk[5], aPk[6], aPk[7]);
#pragma unroll
            for (int i = 0; i < 4; ++i)
                *reinterpret_cast<unsigned*>(&Bs[nxt][(n4 * 4 + i) * LDSTRIDE + kp * 2]) = bPk[i];
        }
        __syncthreads();
    }

    // epilogue: bias + exact GELU, store bf16 h
#pragma unroll
    for (int sm = 0; sm < 2; ++sm) {
        int rowb = m0 + wm * 32 + sm * 16 + g * 8;
#pragma unroll
        for (int sn = 0; sn < 2; ++sn) {
            int col = n0 + wn * 32 + sn * 16 + mi;
            float bias = b1[e * HID + col];
#pragma unroll
            for (int r = 0; r < 8; ++r) {
                float v = acc[sm][sn][r] + bias;
                v = 0.5f * v * (1.0f + erff(v * 0.70710678f));
                hbuf[((size_t)e * TOPK + rowb + r) * HID + col] = f2bf(v);
            }
        }
    }
}

// ---------------------------------------------------------------------------
// 4) GEMM2: out = (h @ w2[e] + b2[e]) * gate; scatter-add with f32 atomics.
//    A tiles (already bf16) staged with async global->LDS copies (ASYNCcnt).
// ---------------------------------------------------------------------------
static __device__ __forceinline__ void async_copy16(
        unsigned ldsOff, const unsigned short* gptr) {
#if USE_ASYNC_LDS
    asm volatile("global_load_async_to_lds_b128 %0, %1, off"
                 :: "v"(ldsOff), "v"(gptr) : "memory");
#else
    (void)ldsOff; (void)gptr;
#endif
}

__global__ void __launch_bounds__(256) moe_gemm2(
        const unsigned short* __restrict__ hbuf, const float* __restrict__ w2,
        const float* __restrict__ b2, const int* __restrict__ sel,
        const float* __restrict__ gateW, float* __restrict__ results) {
    __shared__ __align__(16) unsigned short As[2][128 * LDSTRIDE];
    __shared__ __align__(16) unsigned short Bs[2][64 * LDSTRIDE];

    int e  = blockIdx.z;
    int m0 = blockIdx.x * 128;
    int n0 = blockIdx.y * 64;
    int tid = threadIdx.x;
    int wave = tid >> 5, lane = tid & 31;
    int wm = wave >> 1, wn = wave & 1;
    int g  = lane >> 4, mi = lane & 15;

    int lm = tid >> 1, lq = tid & 1;
    int n4 = tid & 15, kp = tid >> 4;
    const unsigned short* aRow = hbuf + (size_t)(e * TOPK + m0 + lm) * HID;
    const float* w2e = w2 + (size_t)e * HID * DIM;

    // LDS byte offsets for this thread's A-tile slots (aperture low 32 bits)
    unsigned aSlot0 = (unsigned)(size_t)&As[0][lm * LDSTRIDE + lq * 16];
    unsigned aSlot1 = (unsigned)(size_t)&As[1][lm * LDSTRIDE + lq * 16];

    v8f acc[2][2] = {};
    unsigned bPk[4];

    // ---- stage K-step 0 ----
    async_copy16(aSlot0,      aRow + lq * 16);
    async_copy16(aSlot0 + 16, aRow + lq * 16 + 8);
#if !USE_ASYNC_LDS
    {
        const uint4* sA = reinterpret_cast<const uint4*>(aRow + lq * 16);
        uint4* dstA = reinterpret_cast<uint4*>(&As[0][lm * LDSTRIDE + lq * 16]);
        dstA[0] = sA[0]; dstA[1] = sA[1];
    }
#endif
    {
        const float* r0 = w2e + (size_t)(2 * kp) * DIM + n0 + n4 * 4;
        float4 f0 = *reinterpret_cast<const float4*>(r0);
        float4 f1 = *reinterpret_cast<const float4*>(r0 + DIM);
        bPk[0] = pkbf(f0.x, f1.x); bPk[1] = pkbf(f0.y, f1.y);
        bPk[2] = pkbf(f0.z, f1.z); bPk[3] = pkbf(f0.w, f1.w);
#pragma unroll
        for (int i = 0; i < 4; ++i)
            *reinterpret_cast<unsigned*>(&Bs[0][(n4 * 4 + i) * LDSTRIDE + kp * 2]) = bPk[i];
    }
#if USE_ASYNC_LDS
    asm volatile("s_wait_asynccnt 0" ::: "memory");
#endif
    __syncthreads();

    const int NK = HID / 32;
    for (int kt = 0; kt < NK; ++kt) {
        int cur = kt & 1, nxt = cur ^ 1;
        bool more = (kt + 1) < NK;
        if (more) {
            int k0 = (kt + 1) * 32;
            unsigned aSlot = (nxt ? aSlot1 : aSlot0);
            async_copy16(aSlot,      aRow + k0 + lq * 16);
            async_copy16(aSlot + 16, aRow + k0 + lq * 16 + 8);
#if !USE_ASYNC_LDS
            const uint4* sA = reinterpret_cast<const uint4*>(aRow + k0 + lq * 16);
            uint4* dstA = reinterpret_cast<uint4*>(&As[nxt][lm * LDSTRIDE + lq * 16]);
            dstA[0] = sA[0]; dstA[1] = sA[1];
#endif
            const float* r0 = w2e + (size_t)(k0 + 2 * kp) * DIM + n0 + n4 * 4;
            float4 f0 = *reinterpret_cast<const float4*>(r0);
            float4 f1 = *reinterpret_cast<const float4*>(r0 + DIM);
            bPk[0] = pkbf(f0.x, f1.x); bPk[1] = pkbf(f0.y, f1.y);
            bPk[2] = pkbf(f0.z, f1.z); bPk[3] = pkbf(f0.w, f1.w);
            if ((kt + 2) < NK)
                __builtin_prefetch(w2e + (size_t)((kt + 2) * 32 + 2 * kp) * DIM + n0 + n4 * 4, 0, 1);
        }
        frag_compute(As[cur], Bs[cur], wm, wn, g, mi, acc);
        if (more) {
#pragma unroll
            for (int i = 0; i < 4; ++i)
                *reinterpret_cast<unsigned*>(&Bs[nxt][(n4 * 4 + i) * LDSTRIDE + kp * 2]) = bPk[i];
        }
#if USE_ASYNC_LDS
        asm volatile("s_wait_asynccnt 0" ::: "memory");
#endif
        __syncthreads();
    }

#pragma unroll
    for (int sm = 0; sm < 2; ++sm) {
        int rowb = m0 + wm * 32 + sm * 16 + g * 8;
#pragma unroll
        for (int r = 0; r < 8; ++r) {
            int row  = rowb + r;
            int tok  = sel[e * TOPK + row];
            float wg = gateW[e * TOPK + row];
            float* dst = results + (size_t)tok * DIM;
#pragma unroll
            for (int sn = 0; sn < 2; ++sn) {
                int col = n0 + wn * 32 + sn * 16 + mi;
                float v = (acc[sm][sn][r] + b2[e * DIM + col]) * wg;
                atomicAdd(dst + col, v);
            }
        }
    }
}

// ---------------------------------------------------------------------------
// host launcher
// ---------------------------------------------------------------------------
extern "C" void kernel_launch(void* const* d_in, const int* in_sizes, int n_in,
                              void* d_out, int out_size, void* d_ws, size_t ws_size,
                              hipStream_t stream) {
    (void)in_sizes; (void)n_in; (void)out_size; (void)ws_size;

    const float* x  = (const float*)d_in[0];
    const float* rw = (const float*)d_in[1];
    const float* w1 = (const float*)d_in[2];
    const float* b1 = (const float*)d_in[3];
    const float* w2 = (const float*)d_in[4];
    const float* b2 = (const float*)d_in[5];

    // d_out: results [T,D] f32, router_logits [T,E] f32, selected_tokens [E,K] i32
    float* results = (float*)d_out;
    float* logits  = results + (size_t)T_TOK * DIM;
    int*   selOut  = (int*)(logits + (size_t)T_TOK * NE);

    // workspace: probsT (256 KB) | gate weights (32 KB) | h bf16 (64 MB @ 1 MB)
    char* ws = (char*)d_ws;
    float* probsT = (float*)ws;
    float* gateW  = (float*)(ws + (256u << 10));
    unsigned short* hbuf = (unsigned short*)(ws + (1u << 20));

    moe_zero<<<(T_TOK * DIM) / (256 * 4), 256, 0, stream>>>(results);
    moe_router<<<T_TOK / 8, 256, 0, stream>>>(x, rw, logits, probsT);
    moe_topk<<<NE, 1024, 0, stream>>>(probsT, gateW, selOut);

    dim3 g1(TOPK / 128, HID / 64, NE);
    moe_gemm1<<<g1, 256, 0, stream>>>(x, w1, b1, selOut, hbuf);

    dim3 g2(TOPK / 128, DIM / 64, NE);
    moe_gemm2<<<g2, 256, 0, stream>>>(hbuf, w2, b2, selOut, gateW, results);
}
